// IsokawaPytorchLayer_9010841387018
// MI455X (gfx1250) — matrix operand order repaired and checked by reading, not compiled, not run on hardware
//
#include <hip/hip_runtime.h>

// ---------------------------------------------------------------------------
// Isokawa quaternion neuron layer on MI455X (gfx1250, wave32, WMMA).
//
//   out[b, 4n+c] = sigmoid( sum_{m,d} X[b,4m+d] * L(W[n,m])[c,d] - theta[4n+c] )
//
// i.e. a [B x 256] * [256 x 256] GEMM (K = 4*M_IN, N = 4*N_OUT) where the
// 256x256 matrix is the quaternion left-multiplication expansion of W_q.
// Strategy: expand W_q -> f16 matrix in LDS once per block, then f16 WMMA
// (f32 accumulate) over the batch, fused bias-subtract + sigmoid epilogue
// using the hardware v_exp_f32 / v_rcp_f32 transcendentals.
// ---------------------------------------------------------------------------

typedef __attribute__((ext_vector_type(16))) _Float16 v16h;
typedef __attribute__((ext_vector_type(8)))  _Float16 v8h;
typedef __attribute__((ext_vector_type(4)))  _Float16 v4h;
typedef __attribute__((ext_vector_type(8)))  float    v8f;
typedef __attribute__((ext_vector_type(4)))  float    v4f;

#define QK          256   // K = 4 * M_IN
#define QN          256   // N = 4 * N_OUT
#define LDSSTRIDE   264   // QK + 8 halves pad -> kills LDS bank conflicts
#define ROWS_PER_WAVE   16
#define WAVES_PER_BLK   8
#define ROWS_PER_BLK    (ROWS_PER_WAVE * WAVES_PER_BLK)   // 128

__global__ void __launch_bounds__(256)
isokawa_qnn_wmma(const float* __restrict__ X,     // (B, 64, 4) flat
                 const float* __restrict__ Wq,    // (64, 64, 4) flat
                 const float* __restrict__ Th,    // (64, 4) flat
                 float* __restrict__ Out,         // (B, 64, 4) flat
                 long Btot)
{
    // f16 left-mult matrix, stored column-major-by-output: wlds[j][k],
    // j = 4n+c (output column), k = 4m+d (reduction index), k-contiguous.
    __shared__ _Float16 wlds[QN * LDSSTRIDE];     // 135168 B

    const int tid  = threadIdx.x;
    const int lane = tid & 31;
    const int wave = tid >> 5;
    const int lh   = lane & 15;    // row-within-tile (A/C) or column (B/C)
    const int hi   = lane >> 4;    // which half-wave

    const long b0 = (long)blockIdx.x * ROWS_PER_BLK + (long)wave * ROWS_PER_WAVE;
    const float* xrow = X + (b0 + lh) * QK;

    // Overlap the X fetch with the weight expansion below.
    if (b0 < Btot) {
#pragma unroll
        for (int kt = 0; kt < 8; ++kt)
            __builtin_prefetch(xrow + kt * 32, 0, 0);   // global_prefetch_b8
    }

    // ---- Phase 0: expand quaternions into the 256x256 f16 matrix in LDS ----
    // L(q) rows (coefficients over d for output component c):
    //   c=0: ( w, -x, -y, -z)
    //   c=1: ( x,  w, -z,  y)
    //   c=2: ( y,  z,  w, -x)
    //   c=3: ( z, -y,  x,  w)
#pragma unroll
    for (int it = 0; it < 16; ++it) {
        const int q = tid + it * 256;          // quaternion index = n*64 + m
        const int n = q >> 6;
        const int m = q & 63;
        const v4f w = *(const v4f*)(Wq + (size_t)q * 4);
        const _Float16 pw = (_Float16)w.x, px = (_Float16)w.y,
                       py = (_Float16)w.z, pz = (_Float16)w.w;
        const _Float16 nx = -px, ny = -py, nz = -pz;
        const v4h r0 = { pw, nx, ny, nz };
        const v4h r1 = { px, pw, nz, py };
        const v4h r2 = { py, pz, pw, nx };
        const v4h r3 = { pz, ny, px, pw };
        _Float16* dst = &wlds[(size_t)(4 * n) * LDSSTRIDE + 4 * m];
        *(v4h*)(dst)                 = r0;
        *(v4h*)(dst +     LDSSTRIDE) = r1;
        *(v4h*)(dst + 2 * LDSSTRIDE) = r2;
        *(v4h*)(dst + 3 * LDSSTRIDE) = r3;
    }
    __syncthreads();

    if (b0 >= Btot) return;   // wave-uniform; EXEC stays all-ones for WMMA

    // ---- Phase 1: load + convert this wave's 16x256 A tile into fragments --
    // ISA v16h A layout (16x32, MxK): lanes 0-15 row=lane, halves 0-7 ->
    // K=base..base+7, halves 8-15 -> K=base+16..base+23; upper half-wave
    // uses base += 8.
    v16h a[8];
#pragma unroll
    for (int kt = 0; kt < 8; ++kt) {
        const int base = kt * 32 + hi * 8;
        const v4f f0 = *(const v4f*)(xrow + base);
        const v4f f1 = *(const v4f*)(xrow + base + 4);
        const v4f f2 = *(const v4f*)(xrow + base + 16);
        const v4f f3 = *(const v4f*)(xrow + base + 20);
        v16h t;
#pragma unroll
        for (int j = 0; j < 4; ++j) {
            t[j]      = (_Float16)f0[j];
            t[j + 4]  = (_Float16)f1[j];
            t[j + 8]  = (_Float16)f2[j];
            t[j + 12] = (_Float16)f3[j];
        }
        a[kt] = t;
    }

    // ---- GEMM over the 16 N-tiles; 8 chained WMMAs (K=256) per tile --------
    for (int nt = 0; nt < 16; ++nt) {
        v8f acc = {};
        // ISA v16h B layout (32x16, KxN): lane%16 = column, lanes 16-31 get
        // K offset +8, halves 0-7 -> K=base..+7, 8-15 -> K=base+16..+23.
        const _Float16* bcol = &wlds[(size_t)(nt * 16 + lh) * LDSSTRIDE + hi * 8];
#pragma unroll
        for (int kt = 0; kt < 8; ++kt) {
            const v8h blo = *(const v8h*)(bcol + kt * 32);
            const v8h bhi = *(const v8h*)(bcol + kt * 32 + 16);
            v16h bf;
#pragma unroll
            for (int j = 0; j < 8; ++j) { bf[j] = blo[j]; bf[j + 8] = bhi[j]; }
            acc = __builtin_amdgcn_wmma_f32_16x16x32_f16(
                      /*neg_a=*/false, a[kt],
                      /*neg_b=*/false, bf,
                      /*c_mod=*/(short)0, acc,
                      /*reuse_a=*/false, /*reuse_b=*/false);
        }

        // Epilogue: C/D layout -> lane%16 = column, VGPR r = row (+8 upper).
        // sigmoid(s) = rcp(1 + exp2(-s*log2e)) with HW transcendentals:
        //   v_mul + v_exp_f32 + v_add + v_rcp_f32 (TRANS ops co-execute).
        const int   col = nt * 16 + lh;
        const float th  = Th[col];
#pragma unroll
        for (int r = 0; r < 8; ++r) {
            const float s = acc[r] - th;
            const float e = __builtin_amdgcn_exp2f(s * -1.442695040888963f);
            const float y = __builtin_amdgcn_rcpf(1.0f + e);
            Out[(b0 + (long)(hi * 8 + r)) * QN + col] = y;
        }
    }
}

// ---------------------------------------------------------------------------

extern "C" void kernel_launch(void* const* d_in, const int* in_sizes, int n_in,
                              void* d_out, int out_size, void* d_ws, size_t ws_size,
                              hipStream_t stream) {
    const float* X  = (const float*)d_in[0];   // x_batch  (B, 64, 4) f32
    const float* Wq = (const float*)d_in[1];   // W_q      (64, 64, 4) f32
    const float* Th = (const float*)d_in[2];   // theta_q  (64, 4) f32
    float*       O  = (float*)d_out;           // (B, 64, 4) f32

    const long Btot   = (long)in_sizes[0] / QK;            // 131072
    const int  blocks = (int)((Btot + ROWS_PER_BLK - 1) / ROWS_PER_BLK);

    isokawa_qnn_wmma<<<blocks, 256, 0, stream>>>(X, Wq, Th, O, Btot);
}
